// HybridClassifier_90417651515761
// MI455X (gfx1250) — matrix operand order repaired and checked by reading, not compiled, not run on hardware
//
#include <hip/hip_runtime.h>
#include <hip/hip_bf16.h>

// ---------------------------------------------------------------------------
// HybridClassifier for MI455X (gfx1250): f16 WMMA GEMMs + flash attention.
// V=50257 D=512 H=8 NL=4 C=10 CD=64 B=16 L=512 DH=64
// ---------------------------------------------------------------------------

typedef _Float16 v8h  __attribute__((ext_vector_type(8)));
typedef _Float16 v16h __attribute__((ext_vector_type(16)));
typedef float    v8f  __attribute__((ext_vector_type(8)));

#define IDX16 0,1,2,3,4,5,6,7,8,9,10,11,12,13,14,15

// ---------------------------------------------------------------- utilities
__global__ void __launch_bounds__(256) f32_to_f16_kernel(const float* __restrict__ s,
                                                         _Float16* __restrict__ d, int n) {
    int i = blockIdx.x * 256 + threadIdx.x;
    if (i < n) d[i] = (_Float16)s[i];
}

__global__ void __launch_bounds__(256) zero_f32_kernel(float* __restrict__ p, int n) {
    int i = blockIdx.x * 256 + threadIdx.x;
    if (i < n) p[i] = 0.0f;
}

// ------------------------------------------------------------------- embed
// h[b,l,d] = embed[x[b,l],d] + pos[l,d]  (writes f32 master + f16 copy)
__global__ void __launch_bounds__(256) embed_kernel(const int* __restrict__ x,
                                                    const float* __restrict__ emb,
                                                    const float* __restrict__ pos,
                                                    float* __restrict__ h,
                                                    _Float16* __restrict__ hh) {
    int i = blockIdx.x * 256 + threadIdx.x;      // total B*L*D = 4194304
    int d = i & 511;
    int t = i >> 9;                               // b*L + l
    int l = t & 511;
    int tok = x[t];
    float v = emb[(size_t)tok * 512 + d] + pos[(size_t)l * 512 + d];
    h[i] = v;
    hh[i] = (_Float16)v;
}

// ------------------------------------------------------------- NT WMMA GEMM
// Y(N,M) = A(N,K) * W(M,K)^T + bias(M).  One wave -> 16x64 tile.
// EPI: 0 = f32 out, 1 = f16 out, 2 = f16 out with exact GELU.
template <int EPI>
__global__ void __launch_bounds__(256) gemm_nt_wmma(const _Float16* __restrict__ A,
                                                    const _Float16* __restrict__ W,
                                                    const float* __restrict__ bias,
                                                    float* __restrict__ Yf,
                                                    _Float16* __restrict__ Yh,
                                                    int N, int M, int K) {
    int wid  = blockIdx.x * 8 + (threadIdx.x >> 5);
    int lane = threadIdx.x & 31;
    int tilesM = M >> 6;                 // 64-wide column groups
    int tilesN = N >> 4;                 // 16-row tiles
    if (wid >= tilesM * tilesN) return;  // wave-uniform guard (EXEC stays full)
    int tn = wid / tilesM, tm = wid % tilesM;
    int row0 = tn << 4, col0 = tm << 6;
    int r = lane & 15, hf = lane >> 4;

    v8f z = {0.f,0.f,0.f,0.f,0.f,0.f,0.f,0.f};
    v8f acc[4] = {z, z, z, z};

    const _Float16* arow = A + (size_t)(row0 + r) * K + hf * 16;
    for (int k0 = 0; k0 < K; k0 += 32) {
        v8h a1 = *(const v8h*)(arow + k0);
        v8h a2 = *(const v8h*)(arow + k0 + 8);
        v16h af = __builtin_shufflevector(a1, a2, IDX16);
#pragma unroll
        for (int t = 0; t < 4; ++t) {
            const _Float16* bp = W + (size_t)(col0 + t * 16 + r) * K + k0 + hf * 16;
            v16h bf = *(const v16h*)bp;
            acc[t] = __builtin_amdgcn_wmma_f32_16x16x32_f16(
                false, af, false, bf, (short)0, acc[t], false, false);
        }
    }
#pragma unroll
    for (int t = 0; t < 4; ++t) {
        int col = col0 + t * 16 + r;
        float bv = bias ? bias[col] : 0.0f;
#pragma unroll
        for (int j = 0; j < 8; ++j) {
            int row = row0 + hf * 8 + j;              // C/D layout: vgpr j, half hf
            float v = acc[t][j] + bv;
            if (EPI == 2) v = 0.5f * v * (1.0f + erff(v * 0.70710678f));
            if (EPI == 0) Yf[(size_t)row * M + col] = v;
            else          Yh[(size_t)row * M + col] = (_Float16)v;
        }
    }
}

// --------------------------------------------------------- V transpose (f16)
// vT[((b*8+h)*64+dh)*512 + l] = qkv[(b*512+l)*1536 + 1024 + h*64 + dh]
__global__ void __launch_bounds__(256) transpose_v_kernel(const _Float16* __restrict__ qkv,
                                                          _Float16* __restrict__ vT) {
    int i = blockIdx.x * 256 + threadIdx.x;   // total 4194304
    int l  = i & 511;
    int dh = (i >> 9) & 63;
    int h  = (i >> 15) & 7;
    int b  = i >> 18;
    vT[i] = qkv[(size_t)(b * 512 + l) * 1536 + 1024 + h * 64 + dh];
}

// -------------------------------------------------------------- flash attn
// One wave per (b, h, 16-query tile). S = Q K^T (WMMA), online softmax with
// 16-lane xor-shuffle row reductions, P through LDS (D-layout -> A-layout),
// O += P V (WMMA against pre-transposed vT).
__global__ void __launch_bounds__(256) flash_attn_kernel(const _Float16* __restrict__ qkv,
                                                         const _Float16* __restrict__ vT,
                                                         _Float16* __restrict__ oh) {
    __shared__ _Float16 pbuf[8][16 * 32];
    int wslot = threadIdx.x >> 5;
    int lane  = threadIdx.x & 31;
    int wid   = blockIdx.x * 8 + wslot;     // grid sized exactly: 4096 waves
    int qt = wid & 31;
    int h  = (wid >> 5) & 7;
    int b  = wid >> 8;
    int q0 = qt << 4;
    int r = lane & 15, hf = lane >> 4;

    const _Float16* qrow = qkv + (size_t)(b * 512 + q0 + r) * 1536 + h * 64 + hf * 16;
    v16h qa[2];
#pragma unroll
    for (int k = 0; k < 2; ++k) {
        v8h a1 = *(const v8h*)(qrow + k * 32);
        v8h a2 = *(const v8h*)(qrow + k * 32 + 8);
        qa[k] = __builtin_shufflevector(a1, a2, IDX16);
    }

    v8f z = {0.f,0.f,0.f,0.f,0.f,0.f,0.f,0.f};
    v8f acc[4] = {z, z, z, z};
    float rowmax[8], lsum[8], alpha[8];
#pragma unroll
    for (int j = 0; j < 8; ++j) { rowmax[j] = -3.0e38f; lsum[j] = 0.0f; }

    _Float16* pb = pbuf[wslot];

    for (int kt = 0; kt < 512; kt += 32) {
        v8f s0 = z, s1 = z;
#pragma unroll
        for (int k = 0; k < 2; ++k) {
            const _Float16* kp = qkv + (size_t)(b * 512 + kt + r) * 1536 + 512
                                 + h * 64 + k * 32 + hf * 16;
            v16h b0 = *(const v16h*)kp;
            s0 = __builtin_amdgcn_wmma_f32_16x16x32_f16(false, qa[k], false, b0,
                                                        (short)0, s0, false, false);
            v16h b1 = *(const v16h*)(kp + (size_t)16 * 1536);
            s1 = __builtin_amdgcn_wmma_f32_16x16x32_f16(false, qa[k], false, b1,
                                                        (short)0, s1, false, false);
        }
        // online softmax: vgpr j holds row j (lanes 0-15) / row j+8 (lanes 16-31)
#pragma unroll
        for (int j = 0; j < 8; ++j) {
            float a0 = s0[j] * 0.125f, a1 = s1[j] * 0.125f;   // scale = 1/sqrt(64)
            float m = fmaxf(a0, a1);
            m = fmaxf(m, __shfl_xor(m, 1, 32));
            m = fmaxf(m, __shfl_xor(m, 2, 32));
            m = fmaxf(m, __shfl_xor(m, 4, 32));
            m = fmaxf(m, __shfl_xor(m, 8, 32));
            float mn = fmaxf(rowmax[j], m);
            float al = expf(rowmax[j] - mn);
            rowmax[j] = mn;
            float p0 = expf(a0 - mn), p1 = expf(a1 - mn);
            float ps = p0 + p1;
            ps += __shfl_xor(ps, 1, 32);
            ps += __shfl_xor(ps, 2, 32);
            ps += __shfl_xor(ps, 4, 32);
            ps += __shfl_xor(ps, 8, 32);
            lsum[j] = lsum[j] * al + ps;
            alpha[j] = al;
            s0[j] = p0; s1[j] = p1;
        }
#pragma unroll
        for (int t = 0; t < 4; ++t)
#pragma unroll
            for (int j = 0; j < 8; ++j) acc[t][j] *= alpha[j];

        // P (D-layout f32) -> LDS row-major f16 -> A-frag (same wave; DScnt order)
#pragma unroll
        for (int j = 0; j < 8; ++j) {
            int row = j + 8 * hf;
            pb[row * 32 + r]      = (_Float16)s0[j];
            pb[row * 32 + 16 + r] = (_Float16)s1[j];
        }
        v8h p1 = *(const v8h*)(pb + r * 32 + hf * 16);
        v8h p2 = *(const v8h*)(pb + r * 32 + hf * 16 + 8);
        v16h pf = __builtin_shufflevector(p1, p2, IDX16);

#pragma unroll
        for (int t = 0; t < 4; ++t) {
            const _Float16* vp = vT + ((size_t)(b * 8 + h) * 64 + t * 16 + r) * 512
                                 + kt + hf * 16;
            v16h bv = *(const v16h*)vp;
            acc[t] = __builtin_amdgcn_wmma_f32_16x16x32_f16(false, pf, false, bv,
                                                            (short)0, acc[t], false, false);
        }
    }
#pragma unroll
    for (int t = 0; t < 4; ++t)
#pragma unroll
        for (int j = 0; j < 8; ++j) {
            int row = q0 + j + 8 * hf;
            float v = acc[t][j] / lsum[j];
            oh[(size_t)(b * 512 + row) * 512 + h * 64 + t * 16 + r] = (_Float16)v;
        }
}

// --------------------------------------------------------- residual + LN
// h = LN(h + y; g, b); also writes f16 copy. One block per token row, D=512.
__global__ void __launch_bounds__(256) add_ln_kernel(const float* __restrict__ hin,
                                                     const float* __restrict__ y,
                                                     const float* __restrict__ g,
                                                     const float* __restrict__ be,
                                                     float* __restrict__ hout,
                                                     _Float16* __restrict__ hh) {
    int row = blockIdx.x, t = threadIdx.x;
    const float* hp = hin + (size_t)row * 512;
    const float* yp = y + (size_t)row * 512;
    float v0 = hp[t] + yp[t];
    float v1 = hp[t + 256] + yp[t + 256];
    __shared__ float rs[256], rq[256];
    rs[t] = v0 + v1;
    rq[t] = v0 * v0 + v1 * v1;
    __syncthreads();
    for (int s = 128; s > 0; s >>= 1) {
        if (t < s) { rs[t] += rs[t + s]; rq[t] += rq[t + s]; }
        __syncthreads();
    }
    float mean = rs[0] * (1.0f / 512.0f);
    float var  = rq[0] * (1.0f / 512.0f) - mean * mean;
    float rstd = rsqrtf(var + 1e-5f);
    float o0 = (v0 - mean) * rstd * g[t] + be[t];
    float o1 = (v1 - mean) * rstd * g[t + 256] + be[t + 256];
    hout[(size_t)row * 512 + t] = o0;
    hout[(size_t)row * 512 + t + 256] = o1;
    hh[(size_t)row * 512 + t] = (_Float16)o0;
    hh[(size_t)row * 512 + t + 256] = (_Float16)o1;
}

// ----------------------------------------------------------- mean over L
__global__ void __launch_bounds__(256) mean_kernel(const float* __restrict__ h,
                                                   float* __restrict__ ctx) {
    int i = blockIdx.x * 256 + threadIdx.x;   // B*D = 8192
    int b = i >> 9, d = i & 511;
    const float* p = h + (size_t)b * 512 * 512 + d;
    float s = 0.f;
    for (int l = 0; l < 512; ++l) s += p[(size_t)l * 512];
    ctx[i] = s * (1.0f / 512.0f);
}

// ------------------------------------------------------------- histogram
__global__ void __launch_bounds__(256) hist_kernel(const int* __restrict__ x,
                                                   float* __restrict__ counts) {
    int i = blockIdx.x * 256 + threadIdx.x;   // B*L = 8192
    int b = i >> 9;
    int tok = x[i];
    if (tok != 0) atomicAdd(counts + (size_t)b * 50257 + tok, 1.0f);
}

__global__ void __launch_bounds__(256) cnt_stats_kernel(const float* __restrict__ counts,
                                                        float* __restrict__ stats) {
    int b = blockIdx.x, t = threadIdx.x;
    float s = 0.f, q = 0.f;
    for (int v = t; v < 50257; v += 256) {
        float c = counts[(size_t)b * 50257 + v];
        s += c; q += c * c;
    }
    __shared__ float rs[256], rq[256];
    rs[t] = s; rq[t] = q;
    __syncthreads();
    for (int k = 128; k > 0; k >>= 1) {
        if (t < k) { rs[t] += rs[t + k]; rq[t] += rq[t + k]; }
        __syncthreads();
    }
    if (t == 0) {
        float m = rs[0] * (1.0f / 50257.0f);
        float var = rq[0] * (1.0f / 50257.0f) - m * m;
        stats[2 * b] = m;
        stats[2 * b + 1] = rsqrtf(var + 1e-5f);
    }
}

__global__ void __launch_bounds__(256) cnl_kernel(const float* __restrict__ counts,
                                                  const float* __restrict__ stats,
                                                  const float* __restrict__ g,
                                                  const float* __restrict__ be,
                                                  float* __restrict__ cnl, int n) {
    int i = blockIdx.x * 256 + threadIdx.x;   // B*V
    if (i >= n) return;
    int b = i / 50257, v = i % 50257;
    cnl[i] = (counts[i] - stats[2 * b]) * stats[2 * b + 1] * g[v] + be[v];
}

__global__ void __launch_bounds__(256) cf_kernel(const float* __restrict__ cnl,
                                                 const float* __restrict__ Wc,
                                                 const float* __restrict__ bc,
                                                 float* __restrict__ cf) {
    int o = blockIdx.x;              // B*CD = 1024
    int b = o >> 6, m = o & 63, t = threadIdx.x;
    const float* cp = cnl + (size_t)b * 50257;
    const float* wp = Wc + (size_t)m * 50257;
    float s = 0.f;
    for (int v = t; v < 50257; v += 256) s += cp[v] * wp[v];
    __shared__ float rs[256];
    rs[t] = s;
    __syncthreads();
    for (int k = 128; k > 0; k >>= 1) {
        if (t < k) rs[t] += rs[t + k];
        __syncthreads();
    }
    if (t == 0) cf[o] = fmaxf(rs[0] + bc[m], 0.0f);
}

__global__ void __launch_bounds__(256) head_kernel(const float* __restrict__ ctx,
                                                   const float* __restrict__ cf,
                                                   const float* __restrict__ Wh,
                                                   const float* __restrict__ bh,
                                                   float* __restrict__ out) {
    int i = threadIdx.x;
    if (i >= 160) return;            // B*C
    int b = i / 10, c = i % 10;
    const float* w = Wh + (size_t)c * 576;
    const float* cv = ctx + (size_t)b * 512;
    const float* fv = cf + b * 64;
    float s = bh[c];
    for (int d = 0; d < 512; ++d) s += cv[d] * w[d];
    for (int j = 0; j < 64; ++j) s += fv[j] * w[512 + j];
    out[i] = s;
}

// ---------------------------------------------------------------------------
extern "C" void kernel_launch(void* const* d_in, const int* in_sizes, int n_in,
                              void* d_out, int out_size, void* d_ws, size_t ws_size,
                              hipStream_t stream) {
    (void)in_sizes; (void)n_in; (void)out_size; (void)ws_size;
    const int*   x     = (const int*)d_in[0];
    const float* embed = (const float*)d_in[1];
    const float* pos   = (const float*)d_in[2];
    const float* Wqkv  = (const float*)d_in[3];
    const float* bqkv  = (const float*)d_in[4];
    const float* Wo    = (const float*)d_in[5];
    const float* bo    = (const float*)d_in[6];
    const float* ln1_g = (const float*)d_in[7];
    const float* ln1_b = (const float*)d_in[8];
    const float* W1    = (const float*)d_in[9];
    const float* b1    = (const float*)d_in[10];
    const float* W2    = (const float*)d_in[11];
    const float* b2    = (const float*)d_in[12];
    const float* ln2_g = (const float*)d_in[13];
    const float* ln2_b = (const float*)d_in[14];
    const float* cg    = (const float*)d_in[15];
    const float* cb    = (const float*)d_in[16];
    const float* Wc    = (const float*)d_in[17];
    const float* bc    = (const float*)d_in[18];
    const float* Wh    = (const float*)d_in[19];
    const float* bh    = (const float*)d_in[20];
    float* out = (float*)d_out;

    const int NL = 4, Nrow = 16 * 512, Vn = 50257;

    // ------- workspace bump allocator (256B aligned)
    char* base = (char*)d_ws;
    size_t off = 0;
    auto alloc = [&](size_t bytes) -> char* {
        char* p = base + off;
        off = (off + bytes + 255) & ~(size_t)255;
        return p;
    };
    _Float16* Wqkv_h = (_Float16*)alloc((size_t)NL * 1536 * 512 * 2);
    _Float16* Wo_h   = (_Float16*)alloc((size_t)NL * 512 * 512 * 2);
    _Float16* W1_h   = (_Float16*)alloc((size_t)NL * 2048 * 512 * 2);
    _Float16* W2_h   = (_Float16*)alloc((size_t)NL * 512 * 2048 * 2);
    float*    h      = (float*)alloc((size_t)Nrow * 512 * 4);
    _Float16* h_h    = (_Float16*)alloc((size_t)Nrow * 512 * 2);
    _Float16* qkv_h  = (_Float16*)alloc((size_t)Nrow * 1536 * 2);
    _Float16* vT     = (_Float16*)alloc((size_t)16 * 8 * 64 * 512 * 2);
    _Float16* o_h    = (_Float16*)alloc((size_t)Nrow * 512 * 2);
    float*    ybuf   = (float*)alloc((size_t)Nrow * 512 * 4);
    _Float16* ff_h   = (_Float16*)alloc((size_t)Nrow * 2048 * 2);
    float*    counts = (float*)alloc((size_t)16 * Vn * 4);
    float*    cnl    = (float*)alloc((size_t)16 * Vn * 4);
    float*    stats  = (float*)alloc(16 * 2 * 4);
    float*    ctx    = (float*)alloc(16 * 512 * 4);
    float*    cf     = (float*)alloc(16 * 64 * 4);

    // ------- weight conversion to f16
    auto cvt = [&](const float* s, _Float16* d, int n) {
        f32_to_f16_kernel<<<(n + 255) / 256, 256, 0, stream>>>(s, d, n);
    };
    cvt(Wqkv, Wqkv_h, NL * 1536 * 512);
    cvt(Wo,   Wo_h,   NL * 512 * 512);
    cvt(W1,   W1_h,   NL * 2048 * 512);
    cvt(W2,   W2_h,   NL * 512 * 2048);

    // ------- embedding
    embed_kernel<<<(Nrow * 512) / 256, 256, 0, stream>>>(x, embed, pos, h, h_h);

    // ------- encoder layers
    for (int l = 0; l < NL; ++l) {
        // QKV projection -> f16
        {
            int tiles = (Nrow / 16) * (1536 / 64);
            gemm_nt_wmma<1><<<(tiles + 7) / 8, 256, 0, stream>>>(
                h_h, Wqkv_h + (size_t)l * 1536 * 512, bqkv + (size_t)l * 1536,
                nullptr, qkv_h, Nrow, 1536, 512);
        }
        // V transpose for the P*V WMMA B-fragments
        transpose_v_kernel<<<(16 * 8 * 64 * 512) / 256, 256, 0, stream>>>(qkv_h, vT);
        // flash attention: 4096 waves exactly
        flash_attn_kernel<<<512, 256, 0, stream>>>(qkv_h, vT, o_h);
        // output projection -> f32
        {
            int tiles = (Nrow / 16) * (512 / 64);
            gemm_nt_wmma<0><<<(tiles + 7) / 8, 256, 0, stream>>>(
                o_h, Wo_h + (size_t)l * 512 * 512, bo + (size_t)l * 512,
                ybuf, nullptr, Nrow, 512, 512);
        }
        add_ln_kernel<<<Nrow, 256, 0, stream>>>(h, ybuf, ln1_g + (size_t)l * 512,
                                                ln1_b + (size_t)l * 512, h, h_h);
        // FFN up + GELU -> f16
        {
            int tiles = (Nrow / 16) * (2048 / 64);
            gemm_nt_wmma<2><<<(tiles + 7) / 8, 256, 0, stream>>>(
                h_h, W1_h + (size_t)l * 2048 * 512, b1 + (size_t)l * 2048,
                nullptr, ff_h, Nrow, 2048, 512);
        }
        // FFN down -> f32
        {
            int tiles = (Nrow / 16) * (512 / 64);
            gemm_nt_wmma<0><<<(tiles + 7) / 8, 256, 0, stream>>>(
                ff_h, W2_h + (size_t)l * 512 * 2048, b2 + (size_t)l * 512,
                ybuf, nullptr, Nrow, 512, 2048);
        }
        add_ln_kernel<<<Nrow, 256, 0, stream>>>(h, ybuf, ln2_g + (size_t)l * 512,
                                                ln2_b + (size_t)l * 512, h, h_h);
    }

    // ------- context mean
    mean_kernel<<<(16 * 512) / 256, 256, 0, stream>>>(h, ctx);

    // ------- count branch (f32 exact)
    {
        int n = 16 * Vn;
        zero_f32_kernel<<<(n + 255) / 256, 256, 0, stream>>>(counts, n);
        hist_kernel<<<(16 * 512) / 256, 256, 0, stream>>>(x, counts);
        cnt_stats_kernel<<<16, 256, 0, stream>>>(counts, stats);
        cnl_kernel<<<(n + 255) / 256, 256, 0, stream>>>(counts, stats, cg, cb, cnl, n);
        cf_kernel<<<16 * 64, 256, 0, stream>>>(cnl, Wc, bc, cf);
    }

    // ------- classifier head
    head_kernel<<<1, 256, 0, stream>>>(ctx, cf, Wh, bh, out);
}
